// Model_20873541058690
// MI455X (gfx1250) — matrix operand order repaired and checked by reading, not compiled, not run on hardware
//
#include <hip/hip_runtime.h>
#include <math.h>

typedef __attribute__((ext_vector_type(16))) _Float16 v16h;
typedef __attribute__((ext_vector_type(8)))  _Float16 v8h;
typedef __attribute__((ext_vector_type(4)))  _Float16 v4h;
typedef __attribute__((ext_vector_type(8)))  float    v8f;

// Problem sizes (fixed by the reference)
static constexpr int Bc  = 2;
static constexpr int SQ  = 4096;
static constexpr int SKV = 4096;
static constexpr int Hc  = 16;
static constexpr int Dc  = 64;

// Tiling
static constexpr int BM = 128;   // q rows per block (8 waves x 16)
static constexpr int BN = 128;   // kv cols per step (8 16-col tiles)
static constexpr int NW = 8;     // waves per block
static constexpr int NT = NW * 32;
static constexpr int NTile = BN / 16;   // 8 column tiles

__device__ __forceinline__ v16h ld16(const _Float16* p) {
    v8h lo = *(const v8h*)p;
    v8h hi = *(const v8h*)(p + 8);
    return __builtin_shufflevector(lo, hi, 0,1,2,3,4,5,6,7,8,9,10,11,12,13,14,15);
}

// ds_swizzle_b32 XOR-mode lane swap within 32 lanes (imm = {xor[14:10],or[9:5],and[4:0]})
template <int IMM>
__device__ __forceinline__ float fswz(float x) {
    return __int_as_float(__builtin_amdgcn_ds_swizzle(__float_as_int(x), IMM));
}
#define SWZ_X1 0x041F
#define SWZ_X2 0x081F
#define SWZ_X4 0x101F
#define SWZ_X8 0x201F

__global__ __launch_bounds__(NT) void fa_fwd_kernel(const float* __restrict__ Q,
                                                    const float* __restrict__ K,
                                                    const float* __restrict__ V,
                                                    float* __restrict__ O)
{
    // LDS tiles; rows padded by 8 halfs (16B): chunk offsets stay 16B-aligned.
    __shared__ __attribute__((aligned(32))) _Float16 Kt[Dc][BN + 8];     // K^T : [d][kv]
    __shared__ __attribute__((aligned(32))) _Float16 Vt[BN][Dc + 8];     // V   : [kv][d]
    __shared__ __attribute__((aligned(32))) _Float16 Pt[NW][16][BN + 8]; // per-wave P staging

    const int tid  = threadIdx.x;
    const int lane = tid & 31;
    const int wave = tid >> 5;
    const int l16  = lane & 15;
    const int hi   = lane >> 4;          // which 16-lane half

    const int qb = blockIdx.x & ((SQ / BM) - 1);
    const int h  = (blockIdx.x / (SQ / BM)) & (Hc - 1);
    const int b  = blockIdx.x / ((SQ / BM) * Hc);

    const int q0      = qb * BM;
    const int rowbase = q0 + wave * 16;  // first q row owned by this wave
    // exp2-domain scores: fold log2(e) into the Q scale
    const float scale = 0.125f * 1.44269504088896340736f;

    // ---- Q tile for this wave, directly in WMMA A-layout (2 k-steps of 32) ----
    v16h aq[2];
    {
        const int row = rowbase + l16;
        const float* qrow = Q + (((size_t)b * SQ + row) * Hc + h) * Dc;
        #pragma unroll
        for (int c = 0; c < 2; ++c) {
            const int d0 = c * 32 + hi * 8;
            const int d1 = c * 32 + 16 + hi * 8;
            float4 f0 = *(const float4*)(qrow + d0);
            float4 f1 = *(const float4*)(qrow + d0 + 4);
            float4 f2 = *(const float4*)(qrow + d1);
            float4 f3 = *(const float4*)(qrow + d1 + 4);
            float t[16] = { f0.x,f0.y,f0.z,f0.w, f1.x,f1.y,f1.z,f1.w,
                            f2.x,f2.y,f2.z,f2.w, f3.x,f3.y,f3.z,f3.w };
            v16h a;
            #pragma unroll
            for (int i = 0; i < 16; ++i) a[i] = (_Float16)(t[i] * scale);
            aq[c] = a;
        }
    }

    // ---- accumulator state (C/D layout: VGPR r -> row r + hi*8, lane l16 -> col) ----
    v8f zero = {};
    v8f o[4];
    #pragma unroll
    for (int nd = 0; nd < 4; ++nd) o[nd] = zero;
    float m_i[8], l_i[8];
    #pragma unroll
    for (int r = 0; r < 8; ++r) { m_i[r] = -INFINITY; l_i[r] = 0.0f; }

    const int nsteps = qb + 1;          // causal: kv tiles 0 .. q0/BN

    // ---- K/V tile fill: fp32 -> f16, K transposed into LDS ----
    auto fill_tile = [&](int kv0) {
        __syncthreads();                // previous compute done reading LDS
        #pragma unroll
        for (int i = 0; i < (BN * Dc) / (NT * 4); ++i) {   // 8 float4-chunks/thread
            const int idx4 = tid + i * NT;
            const int kv   = idx4 >> 4;                    // 16 float4 per kv row
            const int d4   = (idx4 & 15) << 2;
            const size_t g = (((size_t)b * SKV + (kv0 + kv)) * Hc + h) * Dc + d4;
            float4 kf = *(const float4*)(K + g);
            float4 vf = *(const float4*)(V + g);
            Kt[d4 + 0][kv] = (_Float16)kf.x;               // transpose: scattered b16
            Kt[d4 + 1][kv] = (_Float16)kf.y;
            Kt[d4 + 2][kv] = (_Float16)kf.z;
            Kt[d4 + 3][kv] = (_Float16)kf.w;
            v4h vv = { (_Float16)vf.x, (_Float16)vf.y, (_Float16)vf.z, (_Float16)vf.w };
            *(v4h*)&Vt[kv][d4] = vv;                       // packed b64 store
        }
        __syncthreads();                // tile visible to all waves
    };

    // prefetch next K/V tile lines into cache (global_prefetch_b8, non-blocking)
    auto prefetch_tile = [&](int kv0) {
        #pragma unroll
        for (int i = 0; i < (BN * Dc) / (NT * 4); ++i) {
            const int idx4 = tid + i * NT;
            const int kv   = idx4 >> 4;
            const int d4   = (idx4 & 15) << 2;
            const size_t g = (((size_t)b * SKV + (kv0 + kv)) * Hc + h) * Dc + d4;
            __builtin_prefetch(K + g, 0, 3);
            __builtin_prefetch(V + g, 0, 3);
        }
    };

    // ---- one kv-step: S = Q K^T, online softmax, O += P V.
    // `masked` is a literal at each call site -> two compile-time specializations.
    auto compute_step = [&](int kv0, bool masked) __attribute__((always_inline)) {
        // S = Q * K^T : eight 16-col tiles, K-depth 64 = 2 WMMAs each
        v8f s[NTile];
        #pragma unroll
        for (int nt = 0; nt < NTile; ++nt) {
            v8f c = zero;
            #pragma unroll
            for (int cs = 0; cs < 2; ++cs) {
                v16h bk;
                #pragma unroll
                for (int i = 0; i < 16; ++i)
                    bk[i] = Kt[cs * 32 + (i < 8 ? hi * 8 + i : 16 + hi * 8 + (i - 8))][nt * 16 + l16];
                c = __builtin_amdgcn_wmma_f32_16x16x32_f16(
                        false, aq[cs], false, bk, (short)0, c, false, false);
            }
            s[nt] = c;
        }

        // online softmax (exp2 domain); causal mask only in the masked clone
        float alpha[8];
        #pragma unroll
        for (int r = 0; r < 8; ++r) {
            const int row = rowbase + r + hi * 8;
            float x[NTile];
            #pragma unroll
            for (int nt = 0; nt < NTile; ++nt) {
                float xv = s[nt][r];
                if (masked)
                    xv = (kv0 + nt * 16 + l16 > row) ? -INFINITY : xv;
                x[nt] = xv;
            }
            float rx = fmaxf(fmaxf(fmaxf(x[0], x[1]), fmaxf(x[2], x[3])),
                             fmaxf(fmaxf(x[4], x[5]), fmaxf(x[6], x[7])));
            rx = fmaxf(rx, fswz<SWZ_X1>(rx));
            rx = fmaxf(rx, fswz<SWZ_X2>(rx));
            rx = fmaxf(rx, fswz<SWZ_X4>(rx));
            rx = fmaxf(rx, fswz<SWZ_X8>(rx));
            const float mn = fmaxf(m_i[r], rx);

            _Float16* prow = &Pt[wave][r + hi * 8][0];
            float rs = 0.0f;
            #pragma unroll
            for (int nt = 0; nt < NTile; ++nt) {
                const float p = __builtin_amdgcn_exp2f(x[nt] - mn);
                rs += p;
                prow[nt * 16 + l16] = (_Float16)p;   // stage P for A-layout reload
            }
            rs += fswz<SWZ_X1>(rs);
            rs += fswz<SWZ_X2>(rs);
            rs += fswz<SWZ_X4>(rs);
            rs += fswz<SWZ_X8>(rs);

            const float a = __builtin_amdgcn_exp2f(m_i[r] - mn);
            l_i[r] = l_i[r] * a + rs;
            m_i[r] = mn;
            alpha[r] = a;
        }
        #pragma unroll
        for (int r = 0; r < 8; ++r) {
            const float a = alpha[r];
            o[0][r] *= a; o[1][r] *= a; o[2][r] *= a; o[3][r] *= a;
        }

        // O += P * V : four kv-quarters of 32, A = P in A-layout
        #pragma unroll
        for (int hh = 0; hh < 4; ++hh) {
            v8h plo = *(const v8h*)&Pt[wave][l16][hh * 32 + hi * 8];
            v8h phi = *(const v8h*)&Pt[wave][l16][hh * 32 + 16 + hi * 8];
            v16h pa = __builtin_shufflevector(plo, phi,
                        0,1,2,3,4,5,6,7,8,9,10,11,12,13,14,15);
            #pragma unroll
            for (int nd = 0; nd < 4; ++nd) {
                v16h bv;
                #pragma unroll
                for (int i = 0; i < 16; ++i)
                    bv[i] = Vt[hh * 32 + (i < 8 ? hi * 8 + i : 16 + hi * 8 + (i - 8))][nd * 16 + l16];
                o[nd] = __builtin_amdgcn_wmma_f32_16x16x32_f16(
                            false, pa, false, bv, (short)0, o[nd], false, false);
            }
        }
    };

    // ---- main loop: all steps except the diagonal one are mask-free ----
    for (int step = 0; step < nsteps - 1; ++step) {
        const int kv0 = step * BN;
        fill_tile(kv0);
        prefetch_tile(kv0 + BN);
        compute_step(kv0, false);        // hot path: no compares, no cndmask
    }
    {   // peeled diagonal step
        const int kv0 = (nsteps - 1) * BN;
        fill_tile(kv0);
        compute_step(kv0, true);         // masked clone
    }

    // ---- normalize and write out (coalesced over l16 within each d-chunk) ----
    #pragma unroll
    for (int r = 0; r < 8; ++r) {
        const int row = rowbase + r + hi * 8;
        const float inv = 1.0f / l_i[r];
        float* orow = O + (((size_t)b * SQ + row) * Hc + h) * Dc;
        #pragma unroll
        for (int nd = 0; nd < 4; ++nd)
            orow[nd * 16 + l16] = o[nd][r] * inv;
    }
}

extern "C" void kernel_launch(void* const* d_in, const int* in_sizes, int n_in,
                              void* d_out, int out_size, void* d_ws, size_t ws_size,
                              hipStream_t stream) {
    (void)in_sizes; (void)n_in; (void)out_size; (void)d_ws; (void)ws_size;
    const float* q = (const float*)d_in[0];
    const float* k = (const float*)d_in[1];
    const float* v = (const float*)d_in[2];
    float* out = (float*)d_out;
    const int nblocks = Bc * Hc * (SQ / BM);   // 1024
    fa_fwd_kernel<<<nblocks, NT, 0, stream>>>(q, k, v, out);
}
